// TwoLayerFullPrecisionBPCNN_13881334301020
// MI455X (gfx1250) — matrix operand order repaired and checked by reading, not compiled, not run on hardware
//
#include <hip/hip_runtime.h>
#include <hip/hip_bf16.h>

typedef __attribute__((ext_vector_type(16))) _Float16 v16h;
typedef __attribute__((ext_vector_type(8)))  float    v8f;
typedef __attribute__((ext_vector_type(8)))  unsigned v8u;

namespace {
constexpr int kB = 131072;   // batch
constexpr int kL = 200;      // history length
constexpr int kT = 256;      // table size
constexpr int kLPad = 208;   // 13 * 16 (pad L so K = 2*LPad = 416 = 13 k-tiles of 32)
constexpr int kKTiles = 13;
constexpr int kWavesPerBlock = 8;
constexpr int kThreads = kWavesPerBlock * 32;
constexpr int kTilesTotal = kB / 16;                   // 8192 sample tiles
constexpr int kBlocks = kTilesTotal / kWavesPerBlock;  // 1024

__device__ inline unsigned pack_h2(float x, float y) {
  _Float16 hx = (_Float16)x, hy = (_Float16)y;
  unsigned short ux = __builtin_bit_cast(unsigned short, hx);
  unsigned short uy = __builtin_bit_cast(unsigned short, hy);
  return (unsigned)ux | ((unsigned)uy << 16);
}
} // namespace

__global__ __launch_bounds__(kThreads) void bpcnn_wmma_kernel(
    const int* __restrict__ seq,
    const float* __restrict__ c1_w, const float* __restrict__ c1_b,
    const float* __restrict__ l2_w, const float* __restrict__ l2_b,
    float* __restrict__ out)
{
  // LUT of first layer: lut[t] = packed half2 { tanh(c1_w[0,t]+b0), tanh(c1_w[1,t]+b1) }
  __shared__ __align__(16) unsigned lut[kT];
  // Second-layer weights, filter-interleaved per l: wlut[l] = { l2_w[l], l2_w[L+l] }, 0 for pad
  __shared__ __align__(16) unsigned wlut[kLPad];

  const int tid = threadIdx.x;
  if (tid < kT) {
    float t0 = tanhf(c1_w[tid]      + c1_b[0]);
    float t1 = tanhf(c1_w[kT + tid] + c1_b[1]);
    lut[tid] = pack_h2(t0, t1);
  }
  if (tid < kLPad) {
    float w0 = (tid < kL) ? l2_w[tid]      : 0.0f;
    float w1 = (tid < kL) ? l2_w[kL + tid] : 0.0f;
    wlut[tid] = pack_h2(w0, w1);
  }
  __syncthreads();

  const int wave = tid >> 5;
  const int lane = tid & 31;
  const bool hi  = lane >= 16;                       // lane half-group
  const int tile = blockIdx.x * kWavesPerBlock + wave;
  const int m    = (tile << 4) + (lane & 15);        // sample row for this lane's A data
  const int* row = seq + (size_t)m * kL;

  // K permutation: k = 2*l + f.  A 16x32 f16 layout (ISA 7.12.2):
  //   lanes 0-15 : pairs v=0..3 -> K=2v,2v+1 (l = 16t+v);   v=4..7 -> K=16+.. (l = 16t+8+v-4)
  //   lanes 16-31: pairs v=0..3 -> K=8+2v    (l = 16t+4+v); v=4..7 -> K=24+.. (l = 16t+12+v-4)
  // B 32x16 f16: same K order as A (bv[v] takes wlut at the same l as av[v]).
  const int lofs1 = hi ? 4  : 0;
  const int lofs2 = hi ? 12 : 8;
  const int bofs  = hi ? 8  : 0;

  v8f c = {0.f, 0.f, 0.f, 0.f, 0.f, 0.f, 0.f, 0.f};

  #pragma unroll
  for (int t = 0; t < kKTiles; ++t) {
    const int lb = t * 16;
    // prefetch upcoming part of this row (speculative; harmless at array end)
    __builtin_prefetch(row + lb + 64, 0, 1);

    int4 s1 = make_int4(0, 0, 0, 0);
    int4 s2 = make_int4(0, 0, 0, 0);
    if (lb + lofs1 < kL) s1 = *(const int4*)(row + lb + lofs1);  // aligned: 800B rows, 16B offsets
    if (lb + lofs2 < kL) s2 = *(const int4*)(row + lb + lofs2);

    v8u av, bv;
    av[0] = lut[s1.x & 255];
    av[1] = lut[s1.y & 255];
    av[2] = lut[s1.z & 255];
    av[3] = lut[s1.w & 255];
    av[4] = lut[s2.x & 255];
    av[5] = lut[s2.y & 255];
    av[6] = lut[s2.z & 255];
    av[7] = lut[s2.w & 255];
    #pragma unroll
    for (int v = 0; v < 8; ++v) bv[v] = wlut[lb + (v < 4 ? lofs1 + v : lofs2 + v - 4)];     // broadcast, conflict-free

    v16h a = __builtin_bit_cast(v16h, av);
    v16h b = __builtin_bit_cast(v16h, bv);
    // D = A(16x32 f16) * B(32x16 f16, weights replicated over N) + C(f32)
    c = __builtin_amdgcn_wmma_f32_16x16x32_f16(
        /*neg_a=*/false, a, /*neg_b=*/false, b,
        /*c_mod=*/(short)0, c, /*reuse_a=*/false, /*reuse_b=*/false);
  }

  // C/D layout: VGPR r, lanes 0-15 -> (M=r, N=lane); lanes 16-31 -> (M=8+r, N=lane-16).
  // B replicated over N, so lane 0 holds M=0..7 and lane 16 holds M=8..15 at N=0.
  const float bias = l2_b[0];
  if ((lane & 15) == 0) {
    const int mb = (tile << 4) + (hi ? 8 : 0);
    #pragma unroll
    for (int r = 0; r < 8; ++r) {
      float x = c[r] + bias;
      out[mb + r] = 1.0f / (1.0f + __expf(-x));   // sigmoid
    }
  }
}

extern "C" void kernel_launch(void* const* d_in, const int* in_sizes, int n_in,
                              void* d_out, int out_size, void* d_ws, size_t ws_size,
                              hipStream_t stream) {
  (void)in_sizes; (void)n_in; (void)d_ws; (void)ws_size; (void)out_size;
  const int*   seq  = (const int*)d_in[0];
  const float* c1_w = (const float*)d_in[1];
  const float* c1_b = (const float*)d_in[2];
  const float* l2_w = (const float*)d_in[3];
  const float* l2_b = (const float*)d_in[4];
  float* out = (float*)d_out;

  dim3 grid(kBlocks), block(kThreads);
  hipLaunchKernelGGL(bpcnn_wmma_kernel, grid, block, 0, stream,
                     seq, c1_w, c1_b, l2_w, l2_b, out);
}
